// NonLocalBlock3D_20263655703269
// MI455X (gfx1250) — matrix operand order repaired and checked by reading, not compiled, not run on hardware
//
#include <hip/hip_runtime.h>

// ---------------------------------------------------------------------------
// NonLocalBlock3D for MI455X (gfx1250): bf16 WMMA GEMMs, async-DMA double-
// buffered LDS staging, fp32 GroupNorm/softmax.  B=2, C=512, N=4096.
//
// All GEMMs use B in [N][K] row-major form so both operand tiles are staged
// with contiguous GLOBAL_LOAD_ASYNC_TO_LDS_B128 copies (ASYNCcnt pipeline).
// Scratch (d_ws, ~235 MB):
//   wq/wk/wv/wp bf16   4 * 512*512*2    =   2 MB
//   hfT  bf16 [b][n][c]  2*4096*512*2   =   8 MB
//   qT,kT bf16 [b][n][c], v bf16 [b][c][n] = 24 MB
//   S fp32 [b][i][j]   2*4096*4096*4    = 128 MB
//   P bf16 [b][i][j]   2*4096*4096*2    =  64 MB
//   O2 bf16 [b][i][c]  2*4096*512*2     =   8 MB
// ---------------------------------------------------------------------------

typedef __attribute__((ext_vector_type(16))) __bf16 v16bf;
typedef __attribute__((ext_vector_type(8)))  __bf16 v8bf;
typedef __attribute__((ext_vector_type(8)))  float  v8f;

#define C_DIM   512
#define NPOS    4096
#define BATCHSZ 2
#define NGROUPS 32

// LDS row stride (bf16 elems) for 32-wide K tiles; padded 32->40 so the
// 16-lane fragment gathers (row stride 20 dwords) hit 16 distinct banks.
#define LDS_STRIDE 40
#define LDS_TILE_BYTES (128 * LDS_STRIDE * 2)

// ---------------- CDNA5 async global->LDS copy helpers ---------------------
// Generic pointers to LDS carry the LDS byte offset in their low 32 bits
// (flat->LDS mapping discards the upper bits, ISA 10.2).
static __device__ __forceinline__ unsigned lds_off(const void* p) {
  return (unsigned)(unsigned long long)p;
}
static __device__ __forceinline__ void async_copy_b128(unsigned lds_byte_off,
                                                       const void* gptr) {
  asm volatile("global_load_async_to_lds_b128 %0, %1, off"
               :: "v"(lds_byte_off), "v"((unsigned long long)gptr)
               : "memory");
}
static __device__ __forceinline__ void wait_async0() {
#if __has_builtin(__builtin_amdgcn_s_wait_asynccnt)
  __builtin_amdgcn_s_wait_asynccnt(0);
#else
  asm volatile("s_wait_asynccnt 0x0" ::: "memory");
#endif
}

// ------------------------------- fp32 -> bf16 ------------------------------
__global__ void f32_to_bf16_kernel(const float* __restrict__ src,
                                   __bf16* __restrict__ dst, int n) {
  int i = blockIdx.x * blockDim.x + threadIdx.x;
  if (i < n) dst[i] = (__bf16)src[i];
}

// ------------------------------- GroupNorm ---------------------------------
// One block per (batch, group); writes TRANSPOSED hfT[n][c] (bf16).
__global__ __launch_bounds__(256)
void groupnorm_kernel(const float* __restrict__ x,
                      const float* __restrict__ scale,
                      const float* __restrict__ bias,
                      __bf16* __restrict__ hfT) {
  const int b = blockIdx.x >> 5;     // / NGROUPS
  const int g = blockIdx.x & 31;
  const int chPerG = C_DIM / NGROUPS;          // 16
  const int elems  = chPerG * NPOS;            // 65536
  const float* xb = x + (size_t)b * C_DIM * NPOS + (size_t)g * chPerG * NPOS;
  __bf16* hb = hfT + (size_t)b * C_DIM * NPOS;

  __shared__ float red[256];
  float s = 0.f, ss = 0.f;
  for (int i = threadIdx.x; i < elems; i += 256) {
    float v = xb[i];
    s += v; ss += v * v;
  }
  red[threadIdx.x] = s; __syncthreads();
  for (int off = 128; off > 0; off >>= 1) {
    if ((int)threadIdx.x < off) red[threadIdx.x] += red[threadIdx.x + off];
    __syncthreads();
  }
  const float mean = red[0] * (1.0f / elems);
  __syncthreads();
  red[threadIdx.x] = ss; __syncthreads();
  for (int off = 128; off > 0; off >>= 1) {
    if ((int)threadIdx.x < off) red[threadIdx.x] += red[threadIdx.x + off];
    __syncthreads();
  }
  const float var = red[0] * (1.0f / elems) - mean * mean;
  const float inv = rsqrtf(var + 1e-6f);

  for (int i = threadIdx.x; i < elems; i += 256) {
    const int cl = i >> 12;                    // channel within group
    const int n  = i & (NPOS - 1);
    const int c  = g * chPerG + cl;
    float v = (xb[i] - mean) * inv * scale[c] + bias[c];
    hb[(size_t)n * C_DIM + c] = (__bf16)v;     // transposed store
  }
}

// ------------------------------ WMMA GEMM ----------------------------------
// D = alpha * A[M,K] @ B^T + bias[M] (+ residual).  A row-major [M,K],
// B row-major [N,K] (so D[m,n] = sum_k A[m,k]*B[n,k]).
//   TRANS_OUT: write D as [N,M] (ldc==M), else [M,N] (ldc==N).
// 256 threads (8 waves) per 128x128 tile; wave owns 64x32 (4x2 fragments).
// Double-buffered LDS filled by async global->LDS DMA (ASYNCcnt pipeline).
template<bool TRANS_OUT, typename OutT>
__global__ __launch_bounds__(256)
void wmma_gemm_kernel(const __bf16* __restrict__ A, long long strideA,
                      const __bf16* __restrict__ B, long long strideB,
                      OutT* __restrict__ D, long long strideD,
                      const float* __restrict__ bias,
                      const float* __restrict__ residual, long long strideR,
                      int M, int N, int K, int ldc, float alpha) {
  const int bz = blockIdx.z;
  A += (size_t)bz * strideA;
  B += (size_t)bz * strideB;
  D += (size_t)bz * strideD;

  __shared__ __bf16 lds_a[2][128 * LDS_STRIDE];   // [buf][m][k], padded
  __shared__ __bf16 lds_b[2][128 * LDS_STRIDE];   // [buf][n][k], padded

  const int tid    = threadIdx.x;
  const int lane   = tid & 31;
  const int waveId = tid >> 5;
  const int wm     = (waveId & 1) * 64;        // wave M offset inside tile
  const int wn     = (waveId >> 1) * 32;       // wave N offset inside tile
  const int lrow   = lane & 15;
  const int lhalf  = lane >> 4;
  const int tileN0 = blockIdx.x * 128;
  const int tileM0 = blockIdx.y * 128;

  // staging assignment: thread -> (row, 16-elem half) ; 32B per thread/operand
  const int srow  = tid >> 1;
  const int shalf = tid & 1;
  const unsigned ldsA0 = lds_off(&lds_a[0][0]);
  const unsigned ldsB0 = lds_off(&lds_b[0][0]);
  const unsigned stageOff = (unsigned)(srow * LDS_STRIDE + shalf * 16) * 2u;
  const __bf16* srcA0 = A + (size_t)(tileM0 + srow) * K + shalf * 16;
  const __bf16* srcB0 = B + (size_t)(tileN0 + srow) * K + shalf * 16;

  v8f acc[4][2];
#pragma unroll
  for (int f = 0; f < 4; ++f)
#pragma unroll
    for (int g = 0; g < 2; ++g)
#pragma unroll
      for (int r = 0; r < 8; ++r) acc[f][g][r] = 0.0f;

  // ---- prologue: async-stage tile kt=0 into buffer 0 ----
  {
    const unsigned oa = ldsA0 + stageOff, ob = ldsB0 + stageOff;
    async_copy_b128(oa,      srcA0);
    async_copy_b128(oa + 16, (const char*)srcA0 + 16);
    async_copy_b128(ob,      srcB0);
    async_copy_b128(ob + 16, (const char*)srcB0 + 16);
  }
  wait_async0();
  __syncthreads();

  int cur = 0;
  for (int kt = 0; kt < K; kt += 32) {
    const int nxt = cur ^ 1;
    // ---- prefetch next K-tile into the other buffer (async DMA) ----
    if (kt + 32 < K) {
      const unsigned oa = ldsA0 + (unsigned)nxt * LDS_TILE_BYTES + stageOff;
      const unsigned ob = ldsB0 + (unsigned)nxt * LDS_TILE_BYTES + stageOff;
      const __bf16* sa = srcA0 + kt + 32;
      const __bf16* sb = srcB0 + kt + 32;
      async_copy_b128(oa,      sa);
      async_copy_b128(oa + 16, (const char*)sa + 16);
      async_copy_b128(ob,      sb);
      async_copy_b128(ob + 16, (const char*)sb + 16);
    }

    // ---- build fragments from current buffer (ISA 16-bit lane layouts) ----
    v16bf afrag[4];
#pragma unroll
    for (int f = 0; f < 4; ++f) {
      const __bf16* base =
          &lds_a[cur][(wm + f * 16 + lrow) * LDS_STRIDE + lhalf * 8];
      v8bf lo = *(const v8bf*)base;           // K = lhalf*8 .. +7
      v8bf hi = *(const v8bf*)(base + 16);    // K = 16 + lhalf*8 .. +7
#pragma unroll
      for (int i = 0; i < 8; ++i) { afrag[f][i] = lo[i]; afrag[f][i + 8] = hi[i]; }
    }
    v16bf bfrag[2];
#pragma unroll
    for (int g = 0; g < 2; ++g) {
      const __bf16* base =
          &lds_b[cur][(wn + g * 16 + lrow) * LDS_STRIDE + lhalf * 8];
      v8bf lo = *(const v8bf*)base;           // K = lhalf*8 .. +7
      v8bf hi = *(const v8bf*)(base + 16);    // K = 16 + lhalf*8 .. +7
#pragma unroll
      for (int i = 0; i < 8; ++i) { bfrag[g][i] = lo[i]; bfrag[g][i + 8] = hi[i]; }
    }

#pragma unroll
    for (int f = 0; f < 4; ++f)
#pragma unroll
      for (int g = 0; g < 2; ++g)
        acc[f][g] = __builtin_amdgcn_wmma_f32_16x16x32_bf16(
            false, afrag[f], false, bfrag[g], (short)0, acc[f][g], false, false);

    // my async copies into buf[nxt] done, then WG-wide: everyone finished
    // reading buf[cur] and everyone's copies into buf[nxt] are visible.
    wait_async0();
    __syncthreads();
    cur = nxt;
  }

  // ---- epilogue: alpha, bias, residual, store ----
#pragma unroll
  for (int f = 0; f < 4; ++f)
#pragma unroll
    for (int g = 0; g < 2; ++g)
#pragma unroll
      for (int r = 0; r < 8; ++r) {
        const int m = tileM0 + wm + f * 16 + lhalf * 8 + r;
        const int n = tileN0 + wn + g * 16 + lrow;
        float v = acc[f][g][r] * alpha;
        if (bias) v += bias[m];
        const size_t idx = TRANS_OUT ? ((size_t)n * ldc + m)
                                     : ((size_t)m * ldc + n);
        if (residual) v += residual[(size_t)bz * strideR + idx];
        D[idx] = (OutT)v;
      }
}

// ------------------------------- Softmax -----------------------------------
// One block per attention row (BATCH*NPOS rows of length NPOS).
__global__ __launch_bounds__(256)
void softmax_kernel(const float* __restrict__ S, __bf16* __restrict__ P) {
  const size_t row = blockIdx.x;
  const float* s = S + row * NPOS;
  __bf16* p = P + row * NPOS;
  __shared__ float red[256];

  float loc[16];
  float mx = -3.4e38f;
#pragma unroll
  for (int i = 0; i < 16; ++i) {
    loc[i] = s[threadIdx.x + i * 256];
    mx = fmaxf(mx, loc[i]);
  }
  red[threadIdx.x] = mx; __syncthreads();
  for (int off = 128; off > 0; off >>= 1) {
    if ((int)threadIdx.x < off)
      red[threadIdx.x] = fmaxf(red[threadIdx.x], red[threadIdx.x + off]);
    __syncthreads();
  }
  mx = red[0]; __syncthreads();

  float sum = 0.f;
#pragma unroll
  for (int i = 0; i < 16; ++i) { loc[i] = __expf(loc[i] - mx); sum += loc[i]; }
  red[threadIdx.x] = sum; __syncthreads();
  for (int off = 128; off > 0; off >>= 1) {
    if ((int)threadIdx.x < off) red[threadIdx.x] += red[threadIdx.x + off];
    __syncthreads();
  }
  const float inv = 1.0f / red[0];
#pragma unroll
  for (int i = 0; i < 16; ++i)
    p[threadIdx.x + i * 256] = (__bf16)(loc[i] * inv);
}

// ------------------------------- Launcher ----------------------------------
extern "C" void kernel_launch(void* const* d_in, const int* in_sizes, int n_in,
                              void* d_out, int out_size, void* d_ws, size_t ws_size,
                              hipStream_t stream) {
  const float* x     = (const float*)d_in[0];
  const float* gnsc  = (const float*)d_in[1];
  const float* gnbi  = (const float*)d_in[2];
  const float* wq    = (const float*)d_in[3];
  const float* bq    = (const float*)d_in[4];
  const float* wk    = (const float*)d_in[5];
  const float* bk    = (const float*)d_in[6];
  const float* wv    = (const float*)d_in[7];
  const float* bv    = (const float*)d_in[8];
  const float* wproj = (const float*)d_in[9];
  const float* bproj = (const float*)d_in[10];
  float* out = (float*)d_out;

  // scratch carve-out (256B aligned)
  char* wsp = (char*)d_ws;
  auto carve = [&](size_t bytes) -> char* {
    char* p = wsp;
    wsp += (bytes + 255) & ~(size_t)255;
    return p;
  };
  const size_t CN  = (size_t)C_DIM * NPOS;             // 512*4096
  const size_t NN  = (size_t)NPOS * NPOS;              // 4096*4096
  __bf16* wq_bf = (__bf16*)carve((size_t)C_DIM * C_DIM * 2);
  __bf16* wk_bf = (__bf16*)carve((size_t)C_DIM * C_DIM * 2);
  __bf16* wv_bf = (__bf16*)carve((size_t)C_DIM * C_DIM * 2);
  __bf16* wp_bf = (__bf16*)carve((size_t)C_DIM * C_DIM * 2);
  __bf16* hfT   = (__bf16*)carve(BATCHSZ * CN * 2);    // [b][n][c]
  __bf16* qT    = (__bf16*)carve(BATCHSZ * CN * 2);    // [b][n][c]
  __bf16* kT    = (__bf16*)carve(BATCHSZ * CN * 2);    // [b][n][c]
  __bf16* vbuf  = (__bf16*)carve(BATCHSZ * CN * 2);    // [b][c][n]
  float*  S     = (float*) carve(BATCHSZ * NN * 4);    // [b][i][j]
  __bf16* P     = (__bf16*)carve(BATCHSZ * NN * 2);    // [b][i][j]
  __bf16* O2    = (__bf16*)carve(BATCHSZ * CN * 2);    // [b][i][c]

  const int nw = C_DIM * C_DIM;
  f32_to_bf16_kernel<<<(nw + 255) / 256, 256, 0, stream>>>(wq, wq_bf, nw);
  f32_to_bf16_kernel<<<(nw + 255) / 256, 256, 0, stream>>>(wk, wk_bf, nw);
  f32_to_bf16_kernel<<<(nw + 255) / 256, 256, 0, stream>>>(wv, wv_bf, nw);
  f32_to_bf16_kernel<<<(nw + 255) / 256, 256, 0, stream>>>(wproj, wp_bf, nw);

  groupnorm_kernel<<<BATCHSZ * NGROUPS, 256, 0, stream>>>(x, gnsc, gnbi, hfT);

  // q/k/v projections: D[o,n] = W[o,:] . hfT[n,:].  M=C, N=NPOS, K=C.
  dim3 gQKV(NPOS / 128, C_DIM / 128, BATCHSZ);
  wmma_gemm_kernel<true, __bf16><<<gQKV, 256, 0, stream>>>(   // qT [n][c]
      wq_bf, 0, hfT, (long long)CN, qT, (long long)CN, bq, nullptr, 0,
      C_DIM, NPOS, C_DIM, /*ldc=*/C_DIM, 1.0f);
  wmma_gemm_kernel<true, __bf16><<<gQKV, 256, 0, stream>>>(   // kT [n][c]
      wk_bf, 0, hfT, (long long)CN, kT, (long long)CN, bk, nullptr, 0,
      C_DIM, NPOS, C_DIM, /*ldc=*/C_DIM, 1.0f);
  wmma_gemm_kernel<false, __bf16><<<gQKV, 256, 0, stream>>>(  // v [c][n]
      wv_bf, 0, hfT, (long long)CN, vbuf, (long long)CN, bv, nullptr, 0,
      C_DIM, NPOS, C_DIM, /*ldc=*/NPOS, 1.0f);

  // scores: S[i,j] = (qT[i,:] . kT[j,:]) * C^-1/2.  M=N=NPOS, K=C.
  const float inv_sqrt_c = 0.044194173824159216f;      // 1/sqrt(512)
  dim3 gS(NPOS / 128, NPOS / 128, BATCHSZ);
  wmma_gemm_kernel<false, float><<<gS, 256, 0, stream>>>(
      qT, (long long)CN, kT, (long long)CN, S, (long long)NN,
      nullptr, nullptr, 0, NPOS, NPOS, C_DIM, /*ldc=*/NPOS, inv_sqrt_c);

  softmax_kernel<<<BATCHSZ * NPOS, 256, 0, stream>>>(S, P);

  // O2[i,c] = sum_j P[i,j] * v[c,j].  M=NPOS, N=C, K=NPOS; B=v is [N][K].
  dim3 gO(C_DIM / 128, NPOS / 128, BATCHSZ);
  wmma_gemm_kernel<false, __bf16><<<gO, 256, 0, stream>>>(
      P, (long long)NN, vbuf, (long long)CN, O2, (long long)CN,
      nullptr, nullptr, 0, NPOS, C_DIM, NPOS, /*ldc=*/C_DIM, 1.0f);

  // proj + residual: out[o,n] = x[o,n] + bproj[o] + wproj[o,:] . O2[n,:].
  dim3 gP(NPOS / 128, C_DIM / 128, BATCHSZ);
  wmma_gemm_kernel<false, float><<<gP, 256, 0, stream>>>(
      wp_bf, 0, O2, (long long)CN, out, (long long)CN,
      bproj, x, (long long)CN, C_DIM, NPOS, C_DIM, /*ldc=*/NPOS, 1.0f);

  (void)in_sizes; (void)n_in; (void)out_size; (void)ws_size;
}
